// SimpleRNN_85435489452295
// MI455X (gfx1250) — matrix operand order repaired and checked by reading, not compiled, not run on hardware
//
#include <hip/hip_runtime.h>
#include <hip/hip_bf16.h>
#include <stdint.h>

// ---------------------------------------------------------------------------
// CDNA5 (gfx1250) WMMA types
// ---------------------------------------------------------------------------
typedef __attribute__((ext_vector_type(16))) __bf16 v16bf;
typedef __attribute__((ext_vector_type(8)))  float  v8f;
typedef __attribute__((ext_vector_type(4)))  unsigned int v4u;
typedef __attribute__((ext_vector_type(8)))  int    v8i_;
typedef __attribute__((ext_vector_type(4)))  int    v4i_;

union FragBF {
    uint4 u[2];   // 32 bytes = 16 bf16
    v16bf v;
};

#if defined(__has_builtin)
#  if __has_builtin(__builtin_amdgcn_tensor_load_to_lds)
#    define HAVE_TDM 1
#  else
#    define HAVE_TDM 0
#  endif
#else
#  define HAVE_TDM 0
#endif

// ---------------------------------------------------------------------------
// Async global -> LDS copy (ASYNCcnt path, 16B per lane)
// ---------------------------------------------------------------------------
__device__ __forceinline__ void async_copy16(uint32_t lds_off, uint64_t gaddr) {
    asm volatile("global_load_async_to_lds_b128 %0, %1, off"
                 :: "v"(lds_off), "v"(gaddr) : "memory");
}
__device__ __forceinline__ void wait_async0() {
    asm volatile("s_wait_asynccnt 0" ::: "memory");
}

// ---------------------------------------------------------------------------
// Tensor Data Mover: 2D tile (tile_rows x tile_cols elems, 2B/elem) -> LDS.
// Issued once per workgroup (wave 0); EXEC is ignored by TDM. D# built from
// uniform values per CDNA5 ISA.
// ---------------------------------------------------------------------------
#if HAVE_TDM
__device__ __forceinline__ void tdm_load_tile_bf16(uint32_t lds_off, const void* gptr,
                                                   uint32_t rows_avail, uint32_t cols_avail,
                                                   uint32_t row_stride_elems,
                                                   uint32_t tile_rows, uint32_t tile_cols) {
    uint64_t ga = (uint64_t)gptr;
    v4u g0;
    g0[0] = 1u;                                            // count=1, user mode
    g0[1] = lds_off;                                       // lds_addr
    g0[2] = (uint32_t)(ga & 0xFFFFFFFFu);                  // global_addr[31:0]
    g0[3] = (uint32_t)((ga >> 32) & 0x01FFFFFFu) | 0x80000000u;  // addr[56:32] | type=2
    v8i_ g1;
    g1[0] = (int)(1u << 16);                               // wg_mask=0, data_size=1 (2B)
    g1[1] = (int)((cols_avail & 0xFFFFu) << 16);           // tensor_dim0[15:0] @ bits63:48
    g1[2] = (int)((cols_avail >> 16) | ((rows_avail & 0xFFFFu) << 16)); // dim0 hi | dim1 lo
    g1[3] = (int)((rows_avail >> 16) | (tile_cols << 16)); // dim1 hi | tile_dim0
    g1[4] = (int)tile_rows;                                // tile_dim1 | tile_dim2=0
    g1[5] = (int)row_stride_elems;                         // tensor_dim0_stride[31:0]
    g1[6] = 0;                                             // stride hi | dim1_stride lo
    g1[7] = 0;
    v4i_ z4 = {0, 0, 0, 0};
#if __clang_major__ >= 23
    v8i_ z8 = {0, 0, 0, 0, 0, 0, 0, 0};
    __builtin_amdgcn_tensor_load_to_lds(g0, g1, z4, z4, z8, 0);
#else
    __builtin_amdgcn_tensor_load_to_lds(g0, g1, z4, z4, 0);
#endif
}
#endif

// ---------------------------------------------------------------------------
// f32 -> bf16 elementwise convert
// ---------------------------------------------------------------------------
__global__ void cvt_bf16_kernel(const float* __restrict__ src,
                                __hip_bfloat16* __restrict__ dst, int n) {
    int i = blockIdx.x * blockDim.x + threadIdx.x;
    if (i < n) dst[i] = __float2bfloat16(src[i]);
}

// W_log [128 x 512] row-major  ->  Wlt [512 x 128] bf16
__global__ void cvt_transpose_kernel(const float* __restrict__ src,
                                     __hip_bfloat16* __restrict__ dst) {
    int k = blockIdx.x * blockDim.x + threadIdx.x;   // 0..511
    int n = blockIdx.y;                              // 0..127
    if (k < 512)
        dst[(size_t)k * 128 + n] = __float2bfloat16(src[(size_t)n * 512 + k]);
}

// ---------------------------------------------------------------------------
// Tiled GEMM:  C[M,N] = A[M,K](bf16) * B[K,N](bf16) + bias[N]   (f32 out)
// Block tile 128x64, K-chunk 32; 256 threads = 8 waves. Double-buffered LDS:
//   A tiles staged via per-lane async-to-LDS (ASYNCcnt),
//   B tiles staged via the Tensor Data Mover (TENSORcnt).
// LDS: A0 @0 (8K), A1 @8192 (8K), B0 @16384 (4K), B1 @20480 (4K)
// ---------------------------------------------------------------------------
#define GEMM_LDS_BYTES (2 * 8192 + 2 * 4096)

__global__ __launch_bounds__(256)
void gemm_bf16_wmma(const __hip_bfloat16* __restrict__ A,
                    const __hip_bfloat16* __restrict__ B,
                    const float* __restrict__ bias,
                    float* __restrict__ C,
                    int M, int N, int K) {
    extern __shared__ uint8_t smem[];
    const int tid  = threadIdx.x;
    const int wid  = tid >> 5;
    const int lane = tid & 31;
    const int half = lane >> 4;
    const int l15  = lane & 15;
    const int m0   = blockIdx.y * 128;
    const int n0   = blockIdx.x * 64;

    const uint32_t ldsbase = (uint32_t)(uintptr_t)smem;  // addrspace(3) offset

    auto stageA = [&](int k0, int buf) {
        #pragma unroll
        for (int c = 0; c < 2; ++c) {
            int chunk = tid + c * 256;       // 0..511 : 128 rows x 4 chunks
            int row   = chunk >> 2;
            int co    = chunk & 3;
            uint64_t g = (uint64_t)((const uint8_t*)A +
                         ((size_t)(m0 + row) * K + k0) * 2 + co * 16);
            async_copy16(ldsbase + buf * 8192 + row * 64 + co * 16, g);
        }
    };
    auto stageB = [&](int k0, int buf) {
#if HAVE_TDM
        if (wid == 0) {
            tdm_load_tile_bf16(ldsbase + 16384 + buf * 4096,
                               (const uint8_t*)B + ((size_t)k0 * N + n0) * 2,
                               (uint32_t)(K - k0), (uint32_t)(N - n0),
                               (uint32_t)N, 32u, 64u);
        }
#else
        int row = tid >> 3;                  // 32 rows x 8 chunks of 16B
        int co  = tid & 7;
        uint64_t g = (uint64_t)((const uint8_t*)B +
                     ((size_t)(k0 + row) * N + n0) * 2 + co * 16);
        async_copy16(ldsbase + 16384 + buf * 4096 + row * 128 + co * 16, g);
#endif
    };

    v8f acc[4] = {};

    stageA(0, 0);
    stageB(0, 0);

    const int nIter = K >> 5;
    for (int it = 0; it < nIter; ++it) {
        const int buf = it & 1;
        wait_async0();
#if HAVE_TDM
        if (wid == 0) __builtin_amdgcn_s_wait_tensorcnt(0);
#endif
        __syncthreads();     // staged tiles visible; prev compute done everywhere

        if (it + 1 < nIter) {            // prefetch next chunk into other buffer
            stageA((it + 1) << 5, buf ^ 1);
            stageB((it + 1) << 5, buf ^ 1);
        }

        // ---- A fragment (16x32: row = l15, K-halves by lane/16)
        FragBF af;
        const uint4* arow = (const uint4*)(smem + buf * 8192 + (wid * 16 + l15) * 64);
        af.u[0] = arow[half];
        af.u[1] = arow[2 + half];

        // ---- B fragments (32x16: col = l15, element s = k (s&7)+8*half+16*(s>>3))
        const __bf16* bsm = (const __bf16*)(smem + 16384 + buf * 4096);
        #pragma unroll
        for (int nt = 0; nt < 4; ++nt) {
            FragBF bf2;
            #pragma unroll
            for (int s = 0; s < 16; ++s)
                bf2.v[s] = bsm[((s & 7) + 8 * half + ((s >> 3) << 4)) * 64 + nt * 16 + l15];
            acc[nt] = __builtin_amdgcn_wmma_f32_16x16x32_bf16(
                false, af.v, false, bf2.v, (short)0, acc[nt], false, false);
        }
    }

    // ---- epilogue: D vgpr r -> rows r (lanes 0-15) / r+8 (lanes 16-31)
    #pragma unroll
    for (int nt = 0; nt < 4; ++nt) {
        int col = n0 + nt * 16 + l15;
        float bv = bias[col];
        #pragma unroll
        for (int r = 0; r < 8; ++r) {
            int row = m0 + wid * 16 + half * 8 + r;
            C[(size_t)row * N + col] = acc[nt][r] + bv;
        }
    }
}

// ---------------------------------------------------------------------------
// Sequential relu-RNN scan on ONE persistent workgroup (32 waves, 1 WGP).
// Wave w owns output columns [16w, 16w+16); its 512x16 Wh slice lives in
// VGPRs as 16 WMMA B-fragments. Per step h is broadcast from LDS into the A
// fragment (all 16 rows identical) and the GEMV runs as 16 chained WMMAs.
// ---------------------------------------------------------------------------
__global__ __launch_bounds__(1024)
void rnn_scan_wmma(const float* __restrict__ Xp,            // [T,512] incl bias
                   const __hip_bfloat16* __restrict__ Wh,   // [512,512] bf16
                   const float* __restrict__ h0,            // [512]
                   __hip_bfloat16* __restrict__ Hout,       // [T,512] bf16
                   int T_) {
    __shared__ __align__(16) __hip_bfloat16 hbuf[512];
    const int tid  = threadIdx.x;
    const int w    = tid >> 5;
    const int lane = tid & 31;
    const int half = lane >> 4;
    const int l15  = lane & 15;
    const int col  = w * 16 + l15;

    // Preload Wh B-fragments: chunk j covers K=[32j,32j+32), B col = col, element s = k (s&7)+8*half+16*(s>>3)
    FragBF bfr[16];
    const __bf16* wp = (const __bf16*)Wh;
    #pragma unroll
    for (int j = 0; j < 16; ++j) {
        #pragma unroll
        for (int s = 0; s < 16; ++s)
            bfr[j].v[s] = wp[(size_t)(32 * j + (s & 7) + 8 * half + ((s >> 3) << 4)) * 512 + col];
    }

    if (tid < 512) hbuf[tid] = __float2bfloat16(h0[tid]);
    __syncthreads();

    const uint4* hu4 = (const uint4*)hbuf;   // 4 uint4 per 32-element K-chunk
    const float* xp = Xp + col;
    __hip_bfloat16* ho = Hout + col;

    for (int t = 0; t < T_; ++t) {
        __builtin_prefetch(xp + 16 * 512, 0, 0);   // Xp stream, 16 steps ahead
        float xv = *xp;

        v8f acc = {};
        #pragma unroll
        for (int j = 0; j < 16; ++j) {
            FragBF af;                       // broadcast h chunk into A layout
            af.u[0] = hu4[j * 4 + half];
            af.u[1] = hu4[j * 4 + 2 + half];
            acc = __builtin_amdgcn_wmma_f32_16x16x32_bf16(
                false, af.v, false, bfr[j].v, (short)0, acc, false, false);
        }
        // D row 0 (== all rows since A rows identical); valid in both halves
        float v = acc[0] + xv;
        v = v > 0.f ? v : 0.f;
        __hip_bfloat16 hv = __float2bfloat16(v);

        __syncthreads();                     // all hbuf reads of this step done
        if (lane < 16) {
            hbuf[col] = hv;
            *ho = hv;
        }
        __syncthreads();                     // new h visible

        xp += 512;
        ho += 512;
    }
}

// ---------------------------------------------------------------------------
// Host-side orchestration
// ---------------------------------------------------------------------------
static inline size_t align256(size_t x) { return (x + 255) & ~(size_t)255; }

extern "C" void kernel_launch(void* const* d_in, const int* in_sizes, int n_in,
                              void* d_out, int out_size, void* d_ws, size_t ws_size,
                              hipStream_t stream) {
    (void)in_sizes; (void)n_in; (void)out_size; (void)ws_size;
    const int T = 16384, D = 512, K = 128;

    const float* X    = (const float*)d_in[0];
    const float* Wx0  = (const float*)d_in[1];
    const float* Wh0  = (const float*)d_in[2];
    const float* bh0  = (const float*)d_in[3];
    const float* h00  = (const float*)d_in[4];
    const float* Wx1  = (const float*)d_in[5];
    const float* Wh1  = (const float*)d_in[6];
    const float* bh1  = (const float*)d_in[7];
    const float* h01  = (const float*)d_in[8];
    const float* Wlog = (const float*)d_in[9];
    const float* blog = (const float*)d_in[10];
    float* out = (float*)d_out;

    uint8_t* ws = (uint8_t*)d_ws;
    size_t off = 0;
    auto alloc = [&](size_t bytes) -> void* {
        void* p = ws + off; off += align256(bytes); return p;
    };

    __hip_bfloat16* Xbf  = (__hip_bfloat16*)alloc((size_t)T * D * 2);
    __hip_bfloat16* W0b  = (__hip_bfloat16*)alloc((size_t)D * D * 2);
    __hip_bfloat16* Wh0b = (__hip_bfloat16*)alloc((size_t)D * D * 2);
    __hip_bfloat16* W1b  = (__hip_bfloat16*)alloc((size_t)D * D * 2);
    __hip_bfloat16* Wh1b = (__hip_bfloat16*)alloc((size_t)D * D * 2);
    __hip_bfloat16* Wlt  = (__hip_bfloat16*)alloc((size_t)D * K * 2);
    float*          Xp   = (float*)alloc((size_t)T * D * 4);   // reused layer 0/1
    __hip_bfloat16* H0b  = (__hip_bfloat16*)alloc((size_t)T * D * 2);
    __hip_bfloat16* H1b  = (__hip_bfloat16*)alloc((size_t)T * D * 2);

    // --- convert inputs to bf16
    {
        int n = T * D;
        cvt_bf16_kernel<<<(n + 255) / 256, 256, 0, stream>>>(X, Xbf, n);
        n = D * D;
        cvt_bf16_kernel<<<(n + 255) / 256, 256, 0, stream>>>(Wx0, W0b, n);
        cvt_bf16_kernel<<<(n + 255) / 256, 256, 0, stream>>>(Wh0, Wh0b, n);
        cvt_bf16_kernel<<<(n + 255) / 256, 256, 0, stream>>>(Wx1, W1b, n);
        cvt_bf16_kernel<<<(n + 255) / 256, 256, 0, stream>>>(Wh1, Wh1b, n);
        cvt_transpose_kernel<<<dim3(2, K), 256, 0, stream>>>(Wlog, Wlt);
    }

    // --- layer 0: Xp = X @ Wx0 + bh0 ; scan -> H0 (bf16)
    gemm_bf16_wmma<<<dim3(D / 64, T / 128), 256, GEMM_LDS_BYTES, stream>>>(
        Xbf, W0b, bh0, Xp, T, D, D);
    rnn_scan_wmma<<<1, 1024, 0, stream>>>(Xp, Wh0b, h00, H0b, T);

    // --- layer 1: Xp = H0 @ Wx1 + bh1 ; scan -> H1 (bf16)
    gemm_bf16_wmma<<<dim3(D / 64, T / 128), 256, GEMM_LDS_BYTES, stream>>>(
        H0b, W1b, bh1, Xp, T, D, D);
    rnn_scan_wmma<<<1, 1024, 0, stream>>>(Xp, Wh1b, h01, H1b, T);

    // --- head: out = H1 @ W_log^T + b_log
    gemm_bf16_wmma<<<dim3(K / 64, T / 128), 256, GEMM_LDS_BYTES, stream>>>(
        H1b, Wlt, blog, out, T, K, D);
}
